// QuantLlamaAttention_79697413145083
// MI455X (gfx1250) — matrix operand order repaired and checked by reading, not compiled, not run on hardware
//
#include <hip/hip_runtime.h>
#include <hip/hip_bf16.h>

// ---------------------------------------------------------------------------
// QuantLlamaAttention on MI455X (gfx1250): bf16 WMMA pipeline
//   1) cvt X -> bf16 ; cvt+transpose W_qkv, W_o -> bf16 [N][K]
//   2) GEMM  qkv = X @ Wqkv            (v_wmma_f32_16x16x32_bf16)
//   3) RoPE(neox) + rearrange: Q,K -> [h][s][d], V -> [h][d][s]
//   4) flash attention (causal, online softmax, WMMA for QK^T and PV)
//   5) GEMM  out = attn @ W_o  (fp32 out)
// LDS staging uses GLOBAL_LOAD_ASYNC_TO_LDS_B128 (ASYNCcnt) when available.
// Workspace requirement: 256 MiB.
// ---------------------------------------------------------------------------

typedef __bf16 bf16;
typedef __attribute__((ext_vector_type(16))) __bf16 v16bf;
typedef __attribute__((ext_vector_type(8)))  __bf16 v8bf;
typedef __attribute__((ext_vector_type(8)))  float  v8f;
typedef __attribute__((__vector_size__(16))) int    v4i_vs;   // async-LDS operand type

#define S_LEN 2048
#define HDIM  4096
#define NHEAD 32
#define HEADD 128
#define H3    12288

#if __has_builtin(__builtin_amdgcn_global_load_async_to_lds_b128) && \
    __has_builtin(__builtin_amdgcn_s_wait_asynccnt)
#define USE_ASYNC_LDS 1
#else
#define USE_ASYNC_LDS 0
#endif

__device__ __forceinline__ bf16 f2bf(float f) {
  union { float f; unsigned u; } cv; cv.f = f;
  unsigned r = cv.u + 0x7fffu + ((cv.u >> 16) & 1u);   // round-nearest-even
  union { unsigned short s; bf16 b; } o;
  o.s = (unsigned short)(r >> 16);
  return o.b;
}
__device__ __forceinline__ float bf2f(bf16 b) {
  union { unsigned short s; bf16 b; } i; i.b = b;
  union { unsigned u; float f; } o; o.u = ((unsigned)i.s) << 16;
  return o.f;
}
__device__ __forceinline__ v8f zero8() {
  v8f z = {0.f, 0.f, 0.f, 0.f, 0.f, 0.f, 0.f, 0.f};
  return z;
}

// 16B global -> LDS copy; async DMA (ASYNCcnt) on CDNA5 toolchains that
// declare the builtin, else plain VGPR round-trip.
// Builtin signature (from hipcc diagnostic): param0 is 'v4i __device__*'
// (addrspace(1)); LDS side is the addrspace(3) counterpart.
__device__ __forceinline__ void cp16_g2l(const bf16* g, bf16* l) {
#if USE_ASYNC_LDS
  __builtin_amdgcn_global_load_async_to_lds_b128(
      (__attribute__((address_space(1))) v4i_vs*)(unsigned long long)(g),
      (__attribute__((address_space(3))) v4i_vs*)(l), 0, 0);
#else
  *(v8bf*)l = *(const v8bf*)g;
#endif
}
__device__ __forceinline__ void cp_wait_all() {
#if USE_ASYNC_LDS
  __builtin_amdgcn_s_wait_asynccnt(0);   // all async LDS fills complete
#endif
}

// A-matrix 16x32 bf16 fragment from row-major [>=16][ld] (ISA 7.12.2):
// lanes 0-15: row=lane, K {0..7, 16..23}; lanes 16-31: K {8..15, 24..31}.
__device__ __forceinline__ v16bf load_frag_a(const bf16* base, int ld) {
  const int lane = threadIdx.x & 31;
  const int r    = lane & 15;
  const int koff = (lane >> 4) << 3;                  // 0 or 8
  const bf16* p = base + (size_t)r * ld + koff;
  v8bf lo = *(const v8bf*)p;                          // K = koff .. koff+7
  v8bf hi = *(const v8bf*)(p + 16);                   // K = koff+16 .. koff+23
  v16bf f;
#pragma unroll
  for (int i = 0; i < 8; ++i) { f[i] = lo[i]; f[i + 8] = hi[i]; }
  return f;
}

// B-matrix 32x16 fragment loaded from B^T stored row-major [N][K]:
// lane = column N; lanes 0-15 hold K {0..7, 16..23}, lanes 16-31 hold K {8..15, 24..31}.
__device__ __forceinline__ v16bf load_frag_b(const bf16* baseT, int ld) {
  const int lane = threadIdx.x & 31;
  const int n    = lane & 15;
  const int koff = (lane >> 4) << 3;                  // 0 or 8
  const bf16* p = baseT + (size_t)n * ld + koff;
  v8bf lo = *(const v8bf*)p;
  v8bf hi = *(const v8bf*)(p + 16);
  v16bf f;
#pragma unroll
  for (int i = 0; i < 8; ++i) { f[i] = lo[i]; f[i + 8] = hi[i]; }
  return f;
}

// ------------------------------ pre-pass -----------------------------------

__global__ void k_cvt(const float* __restrict__ x, bf16* __restrict__ y, int n) {
  int i = blockIdx.x * blockDim.x + threadIdx.x;
  if (i < n) y[i] = f2bf(x[i]);
}

// W[K][N] fp32 -> WT[N][K] bf16 (LDS-tiled 32x32 transpose)
__global__ __launch_bounds__(256)
void k_transpose_cvt(const float* __restrict__ W, bf16* __restrict__ WT,
                     int K, int N) {
  __shared__ float tile[32][33];
  const int n0 = blockIdx.x * 32, k0 = blockIdx.y * 32;
  const int tx = threadIdx.x & 31, ty = threadIdx.x >> 5;   // 32 x 8
#pragma unroll
  for (int j = 0; j < 4; ++j)
    tile[ty + j * 8][tx] = W[(size_t)(k0 + ty + j * 8) * N + n0 + tx];
  __syncthreads();
#pragma unroll
  for (int j = 0; j < 4; ++j)
    WT[(size_t)(n0 + ty + j * 8) * K + k0 + tx] = f2bf(tile[tx][ty + j * 8]);
}

// ------------------------------- GEMM --------------------------------------
// C[M][N] = A[M][K] * BT[N][K]^T ; 128x128x32 tiles, 8 waves,
// double-buffered LDS filled by async global->LDS DMA.
template <bool OUT_BF16>
__global__ __launch_bounds__(256)
void k_gemm(const bf16* __restrict__ A, const bf16* __restrict__ BT,
            void* __restrict__ Cout, int M, int N, int K) {
  __shared__ bf16 sA[2][128][40];   // padded: 80B rows, 16B aligned
  __shared__ bf16 sB[2][128][40];

  const int tid  = threadIdx.x;
  const int wave = tid >> 5;
  const int wm   = wave & 3;                // 4 waves x 32 rows
  const int wn   = wave >> 2;               // 2 waves x 64 cols
  const int m0   = blockIdx.y * 128;
  const int n0   = blockIdx.x * 128;

  const int lrow = tid >> 1;                // 0..127
  const int lcol = (tid & 1) << 4;          // 0 or 16
  const bf16* gA = A  + (size_t)(m0 + lrow) * K + lcol;
  const bf16* gB = BT + (size_t)(n0 + lrow) * K + lcol;

  v8f acc[2][4];
#pragma unroll
  for (int i = 0; i < 2; ++i)
#pragma unroll
    for (int j = 0; j < 4; ++j) acc[i][j] = zero8();

  const int nk = K >> 5;
  // stage tile 0 (async DMA straight into LDS)
  cp16_g2l(gA,     &sA[0][lrow][lcol]);
  cp16_g2l(gA + 8, &sA[0][lrow][lcol + 8]);
  cp16_g2l(gB,     &sB[0][lrow][lcol]);
  cp16_g2l(gB + 8, &sB[0][lrow][lcol + 8]);
  cp_wait_all();
  __syncthreads();
  int buf = 0;

  for (int kt = 0; kt < nk; ++kt) {
    if (kt + 1 < nk) {                      // kick off next K-slab DMA
      const bf16* pA = gA + (size_t)(kt + 1) * 32;
      const bf16* pB = gB + (size_t)(kt + 1) * 32;
      cp16_g2l(pA,     &sA[buf ^ 1][lrow][lcol]);
      cp16_g2l(pA + 8, &sA[buf ^ 1][lrow][lcol + 8]);
      cp16_g2l(pB,     &sB[buf ^ 1][lrow][lcol]);
      cp16_g2l(pB + 8, &sB[buf ^ 1][lrow][lcol + 8]);
    }
    v16bf af0 = load_frag_a(&sA[buf][wm * 32][0], 40);
    v16bf af1 = load_frag_a(&sA[buf][wm * 32 + 16][0], 40);
#pragma unroll
    for (int ni = 0; ni < 4; ++ni) {
      v16bf bfm = load_frag_b(&sB[buf][wn * 64 + ni * 16][0], 40);
      acc[0][ni] = __builtin_amdgcn_wmma_f32_16x16x32_bf16(
          false, af0, false, bfm, (short)0, acc[0][ni], false, false);
      acc[1][ni] = __builtin_amdgcn_wmma_f32_16x16x32_bf16(
          false, af1, false, bfm, (short)0, acc[1][ni], false, false);
    }
    if (kt + 1 < nk) {
      cp_wait_all();                        // DMA into buf^1 complete
      __syncthreads();                      // (also fences this buf's reads)
      buf ^= 1;
    }
  }

  // epilogue: C/D layout — lanes 0-15: M=j, N=lane; lanes 16-31: M=j+8
  const int lane = tid & 31;
  const int cl   = lane & 15;
  const int rhi  = (lane >> 4) << 3;
#pragma unroll
  for (int mi = 0; mi < 2; ++mi)
#pragma unroll
    for (int ni = 0; ni < 4; ++ni) {
      const int row0 = m0 + wm * 32 + mi * 16 + rhi;
      const int col  = n0 + wn * 64 + ni * 16 + cl;
#pragma unroll
      for (int j = 0; j < 8; ++j) {
        const size_t idx = (size_t)(row0 + j) * N + col;
        if (OUT_BF16) ((bf16*)Cout)[idx]  = f2bf(acc[mi][ni][j]);
        else          ((float*)Cout)[idx] = acc[mi][ni][j];
      }
    }
}

// --------------------------- RoPE + rearrange -------------------------------
// qkv bf16 [S][12288] -> Q,K (rotated) [h][s][d] ; V^T [h][d][s]
__global__ void k_rope(const bf16* __restrict__ QKV, const int* __restrict__ pos,
                       bf16* __restrict__ QO, bf16* __restrict__ KO,
                       bf16* __restrict__ VTO) {
  const int s = blockIdx.x;
  const int h = blockIdx.y;
  const int d = threadIdx.x;                         // 0..63 (half dim)
  const float inv_freq = exp2f(-(float)d * 0.2076205059304601f); // log2(1e4)/64
  const float ang = (float)pos[s] * inv_freq;
  const float cs = cosf(ang), sn = sinf(ang);

  const size_t rb = (size_t)s * H3 + (size_t)h * HEADD;
  const float q1 = bf2f(QKV[rb + d]);
  const float q2 = bf2f(QKV[rb + d + 64]);
  const float k1 = bf2f(QKV[rb + HDIM + d]);
  const float k2 = bf2f(QKV[rb + HDIM + d + 64]);
  const float v1 = bf2f(QKV[rb + 2 * HDIM + d]);
  const float v2 = bf2f(QKV[rb + 2 * HDIM + d + 64]);

  const size_t ob = ((size_t)h * S_LEN + s) * HEADD;
  QO[ob + d]      = f2bf(q1 * cs - q2 * sn);
  QO[ob + d + 64] = f2bf(q2 * cs + q1 * sn);
  KO[ob + d]      = f2bf(k1 * cs - k2 * sn);
  KO[ob + d + 64] = f2bf(k2 * cs + k1 * sn);
  VTO[((size_t)h * HEADD + d) * S_LEN + s]      = f2bf(v1);
  VTO[((size_t)h * HEADD + d + 64) * S_LEN + s] = f2bf(v2);
}

// ---------------------------- flash attention -------------------------------
// block = (head, 64-query tile), 4 waves x 16 query rows, 64-key tiles.
__global__ __launch_bounds__(128)
void k_attn(const bf16* __restrict__ QB, const bf16* __restrict__ KB,
            const bf16* __restrict__ VT, bf16* __restrict__ OUT) {
  __shared__ bf16 sK[64][136];       // K tile  [key][d]   (= B^T for QK^T)
  __shared__ bf16 sV[128][72];       // V^T tile [d][key]  (= B^T for PV)
  __shared__ bf16 sP[4][16][72];     // per-wave P strip (D-layout -> A-layout)

  const int h    = blockIdx.y;
  const int qt   = blockIdx.x;
  const int tid  = threadIdx.x;
  const int wave = tid >> 5;
  const int lane = tid & 31;
  const int cl   = lane & 15;
  const int rhi  = (lane >> 4) << 3;
  const int q0   = qt * 64 + wave * 16;

  // Q fragments held in registers for the whole block (4 x 32 of head_dim)
  const bf16* qbase = QB + ((size_t)h * S_LEN + q0) * HEADD;
  v16bf qf[4];
#pragma unroll
  for (int kk = 0; kk < 4; ++kk) qf[kk] = load_frag_a(qbase + kk * 32, HEADD);

  v8f oacc[8];
#pragma unroll
  for (int i = 0; i < 8; ++i) oacc[i] = zero8();
  float mrow[8], lrow[8];
#pragma unroll
  for (int j = 0; j < 8; ++j) { mrow[j] = -1e30f; lrow[j] = 0.f; }

  const float c2 = 0.08838834764831845f * 1.4426950408889634f; // scale*log2e

  for (int kt = 0; kt <= qt; ++kt) {
    const int k0 = kt * 64;
    {   // stage K tile (64x128) and V^T tile (128x64) via async LDS DMA
      const int row = tid >> 1, seg = (tid & 1) * 64;
      const bf16* ksrc = KB + ((size_t)h * S_LEN + k0 + row) * HEADD + seg;
#pragma unroll
      for (int i = 0; i < 8; ++i)
        cp16_g2l(ksrc + i * 8, &sK[row][seg + i * 8]);
      const bf16* vsrc = VT + ((size_t)h * HEADD + tid) * S_LEN + k0;
#pragma unroll
      for (int i = 0; i < 8; ++i)
        cp16_g2l(vsrc + i * 8, &sV[tid][i * 8]);
    }
    cp_wait_all();
    __syncthreads();

    // scores: 16q x 64k
    v8f sacc[4];
#pragma unroll
    for (int c = 0; c < 4; ++c) sacc[c] = zero8();
#pragma unroll
    for (int c = 0; c < 4; ++c)
#pragma unroll
      for (int kk = 0; kk < 4; ++kk) {
        v16bf bfk = load_frag_b(&sK[c * 16][kk * 32], 136);
        sacc[c] = __builtin_amdgcn_wmma_f32_16x16x32_bf16(
            false, qf[kk], false, bfk, (short)0, sacc[c], false, false);
      }

    if (kt == qt) {                         // causal mask on diagonal tile
#pragma unroll
      for (int c = 0; c < 4; ++c)
#pragma unroll
        for (int j = 0; j < 8; ++j)
          if (k0 + c * 16 + cl > q0 + rhi + j) sacc[c][j] = -1e30f;
    }

    // online softmax (rows: VGPR slot j x lane-half; cols: 16-lane groups)
#pragma unroll
    for (int j = 0; j < 8; ++j) {
      float v = fmaxf(fmaxf(sacc[0][j], sacc[1][j]),
                      fmaxf(sacc[2][j], sacc[3][j]));
      v = fmaxf(v, __shfl_xor(v, 1));
      v = fmaxf(v, __shfl_xor(v, 2));
      v = fmaxf(v, __shfl_xor(v, 4));
      v = fmaxf(v, __shfl_xor(v, 8));
      const float mn    = fmaxf(mrow[j], v);
      const float alpha = exp2f((mrow[j] - mn) * c2);
      mrow[j] = mn;
      lrow[j] *= alpha;
#pragma unroll
      for (int ni = 0; ni < 8; ++ni) oacc[ni][j] *= alpha;
      float ps = 0.f;
#pragma unroll
      for (int c = 0; c < 4; ++c) {
        const float p = exp2f((sacc[c][j] - mn) * c2);
        ps += p;
        sP[wave][rhi + j][c * 16 + cl] = f2bf(p);
      }
      ps += __shfl_xor(ps, 1);
      ps += __shfl_xor(ps, 2);
      ps += __shfl_xor(ps, 4);
      ps += __shfl_xor(ps, 8);
      lrow[j] += ps;
    }

    // P @ V : accumulate 16q x 128d
#pragma unroll
    for (int kkp = 0; kkp < 2; ++kkp) {
      v16bf pf = load_frag_a(&sP[wave][0][kkp * 32], 72);
#pragma unroll
      for (int ni = 0; ni < 8; ++ni) {
        v16bf vf = load_frag_b(&sV[ni * 16][kkp * 32], 72);
        oacc[ni] = __builtin_amdgcn_wmma_f32_16x16x32_bf16(
            false, pf, false, vf, (short)0, oacc[ni], false, false);
      }
    }
    __syncthreads();
  }

  // normalize and write attn output as bf16 [s][H] (A operand of GEMM2)
#pragma unroll
  for (int ni = 0; ni < 8; ++ni)
#pragma unroll
    for (int j = 0; j < 8; ++j) {
      const int s = q0 + rhi + j;
      const int d = ni * 16 + cl;
      OUT[(size_t)s * HDIM + (size_t)h * HEADD + d] =
          f2bf(oacc[ni][j] / lrow[j]);
    }
}

// ------------------------------- launch -------------------------------------

extern "C" void kernel_launch(void* const* d_in, const int* in_sizes, int n_in,
                              void* d_out, int out_size, void* d_ws, size_t ws_size,
                              hipStream_t stream) {
  const float* hidden = (const float*)d_in[0];
  const int*   pos    = (const int*)d_in[1];
  const float* Wqkv   = (const float*)d_in[2];
  const float* Wo     = (const float*)d_in[3];

  char* ws = (char*)d_ws;
  size_t o = 0;
  bf16* XB    = (bf16*)(ws + o); o += (size_t)S_LEN * HDIM * 2;   // 16 MiB
  bf16* WQKVT = (bf16*)(ws + o); o += (size_t)HDIM * H3 * 2;      // 96 MiB
  bf16* WOT   = (bf16*)(ws + o); o += (size_t)HDIM * HDIM * 2;    // 32 MiB
  bf16* QKVB  = (bf16*)(ws + o); o += (size_t)S_LEN * H3 * 2;     // 48 MiB
  bf16* QB    = (bf16*)(ws + o); o += (size_t)S_LEN * HDIM * 2;   // 16 MiB
  bf16* KB    = (bf16*)(ws + o); o += (size_t)S_LEN * HDIM * 2;   // 16 MiB
  bf16* VTB   = (bf16*)(ws + o); o += (size_t)S_LEN * HDIM * 2;   // 16 MiB
  bf16* ATTNB = (bf16*)(ws + o); o += (size_t)S_LEN * HDIM * 2;   // 16 MiB

  k_cvt<<<(S_LEN * HDIM) / 256, 256, 0, stream>>>(hidden, XB, S_LEN * HDIM);
  k_transpose_cvt<<<dim3(H3 / 32, HDIM / 32), 256, 0, stream>>>(Wqkv, WQKVT, HDIM, H3);
  k_transpose_cvt<<<dim3(HDIM / 32, HDIM / 32), 256, 0, stream>>>(Wo, WOT, HDIM, HDIM);

  k_gemm<true><<<dim3(H3 / 128, S_LEN / 128), 256, 0, stream>>>(
      XB, WQKVT, QKVB, S_LEN, H3, HDIM);

  k_rope<<<dim3(S_LEN, NHEAD), 64, 0, stream>>>(QKVB, pos, QB, KB, VTB);

  k_attn<<<dim3(S_LEN / 64, NHEAD), 128, 0, stream>>>(QB, KB, VTB, ATTNB);

  k_gemm<false><<<dim3(HDIM / 128, S_LEN / 128), 256, 0, stream>>>(
      ATTNB, WOT, d_out, S_LEN, HDIM, HDIM);
}
